// MC_E_GCL_91328184582715
// MI455X (gfx1250) — matrix operand order repaired and checked by reading, not compiled, not run on hardware
//
#include <hip/hip_runtime.h>
#include <cstddef>

typedef __attribute__((ext_vector_type(16))) __bf16 v16bf;
typedef __attribute__((ext_vector_type(8)))  __bf16 v8bf;
typedef __attribute__((ext_vector_type(8)))  float  v8f;

__device__ __forceinline__ float silu_f(float x) {
    return x / (1.0f + __expf(-x));
}

__device__ __forceinline__ v8f wmma_bf16(v16bf a, v16bf b, v8f c) {
    // D = A(16x32 bf16) * B(32x16 bf16) + C(16x16 f32)
    return __builtin_amdgcn_wmma_f32_16x16x32_bf16(
        false, a, false, b, (short)0, c, false, false);
}

// A-fragment (16x32 bf16): lane L (m = L%16, g = L/16) holds
// K = g*8 + [0..7] in VGPRs 0-3 and K = 16 + g*8 + [0..7] in VGPRs 4-7.
__device__ __forceinline__ v16bf load_a_frag(const __bf16* p) {
    v16bf a;
    ((v8bf*)&a)[0] = *(const v8bf*)(p);
    ((v8bf*)&a)[1] = *(const v8bf*)(p + 16);
    return a;
}

// ---------------------------------------------------------------------------
// Weight pre-swizzle: W [Ksrc x Nsrc] row-major f32 -> bf16 B-fragment layout.
// B-fragment (32x16): lane L (n = L%16, g = L/16) holds K = g*8 + j (j<8), 16 + g*8 + (j-8) (j>=8).
// packed[((kt*Nt + nt)*32 + lane)*16 + j] = W[kt*32 + g*8 + (j&7) + (j>>3)*16][nt*16 + n]
// ---------------------------------------------------------------------------
__global__ void pack_weight_kernel(const float* __restrict__ W,
                                   __bf16* __restrict__ out,
                                   int Kpad, int Ksrc, int Npad, int Nsrc) {
    int tid = blockIdx.x * blockDim.x + threadIdx.x;
    int total = Kpad * Npad;
    if (tid >= total) return;
    int j    = tid & 15;
    int lane = (tid >> 4) & 31;
    int tile = tid >> 9;
    int Nt   = Npad >> 4;
    int nt   = tile % Nt;
    int kt   = tile / Nt;
    int k = kt * 32 + (lane >> 4) * 8 + (j & 7) + (j >> 3) * 16;
    int n = nt * 16 + (lane & 15);
    float v = (k < Ksrc && n < Nsrc) ? W[(size_t)k * Nsrc + n] : 0.0f;
    out[tid] = (__bf16)v;
}

__global__ void zero_kernel(float* __restrict__ p, long long n) {
    long long i = (long long)blockIdx.x * blockDim.x + threadIdx.x;
    if (i < n) p[i] = 0.0f;
}

// ---------------------------------------------------------------------------
// Fused edge kernel: 64 edges per block, 256 threads (8 wave32).
//   e_in = [h[row] | h[col] | radial | 0]  (K = 288)
//   m1 = silu(e_in @ We1 + be1); m2 = silu(m1 @ We2 + be2)
//   agg[row]  += m2                           (f32 atomics)
//   cw = silu(m2 @ Wc1 + bc1) @ Wc2 (N padded 2->16)
//   agg_c[row] += coord_diff * cw ; cnt[row] += 1
// ---------------------------------------------------------------------------
__global__ __launch_bounds__(256, 1)
void egcl_edge_kernel(const float* __restrict__ h,
                      const int*   __restrict__ eidx,
                      const float* __restrict__ coord,
                      const float* __restrict__ be1,
                      const float* __restrict__ be2,
                      const float* __restrict__ bc1,
                      const __bf16* __restrict__ We1p,
                      const __bf16* __restrict__ We2p,
                      const __bf16* __restrict__ Wc1p,
                      const __bf16* __restrict__ Wc2p,
                      float* __restrict__ agg,
                      float* __restrict__ aggc,
                      float* __restrict__ cnt,
                      int Nn, int Ne)
{
    // lds_big: e_in tile [64][296] bf16; reused as m2 [64][136] after layer 1.
    // lds_m:   m1 [64][136] bf16; reused as p (coord hidden) after layer 2.
    __shared__ __bf16 lds_big[64 * 296];
    __shared__ __bf16 lds_m[64 * 136];
    __shared__ int   s_row[64];
    __shared__ int   s_col[64];
    __shared__ float s_cd[64 * 6];   // coord_diff [e][c][xyz]

    const int t    = threadIdx.x;
    const int w    = t >> 5;
    const int lane = t & 31;
    const int g    = lane >> 4;
    const int ln   = lane & 15;
    const int ebase = blockIdx.x * 64;
    if (ebase >= Ne) return;

    // ---------------- Phase A: stage indices, h gather, radial ----------------
    if (t < 64) {
        int e = ebase + t;
        if (e >= Ne) e = Ne - 1;      // duplicate-clamp tail (E % 64 == 0 here)
        s_row[t] = eidx[e];
        s_col[t] = eidx[Ne + e];
    }
    __syncthreads();

    {   // 4 threads per edge row; each stages 64 contiguous features
        int e = t >> 2, p = t & 3;
        int src = (p < 2) ? s_row[e] : s_col[e];
        const float4* hp = (const float4*)(h + (size_t)src * 128 + (p & 1) * 64);
        uint4* dst = (uint4*)(lds_big + e * 296 + p * 64);
        #pragma unroll
        for (int i = 0; i < 8; ++i) {
            float4 x = hp[2 * i];
            float4 y = hp[2 * i + 1];
            union { __bf16 b[8]; uint4 u; } pk;
            pk.b[0] = (__bf16)x.x; pk.b[1] = (__bf16)x.y;
            pk.b[2] = (__bf16)x.z; pk.b[3] = (__bf16)x.w;
            pk.b[4] = (__bf16)y.x; pk.b[5] = (__bf16)y.y;
            pk.b[6] = (__bf16)y.z; pk.b[7] = (__bf16)y.w;
            dst[i] = pk.u;
        }
    }
    if (t < 64) {   // radial features + coord_diff, one thread per edge
        int r = s_row[t], c = s_col[t];
        const float* cr = coord + (size_t)r * 6;
        const float* cc = coord + (size_t)c * 6;
        float da[3], s0[3], s1[3], cd1[3];
        float dd = 0.f, d0 = 0.f, d1 = 0.f, p0 = 0.f, p1 = 0.f, p01 = 0.f;
        #pragma unroll
        for (int k = 0; k < 3; ++k) {
            float ci0 = cr[k], ci1 = cr[3 + k];
            float cj0 = cc[k], cj1 = cc[3 + k];
            da[k]  = ci0 - cj0;
            s0[k]  = ci1 - ci0;
            s1[k]  = cj1 - cj0;
            cd1[k] = ci1 - cj1;
            dd  += da[k] * da[k];  d0 += s0[k] * s0[k];  d1 += s1[k] * s1[k];
            p0  += s0[k] * da[k];  p1 += s1[k] * da[k];  p01 += s0[k] * s1[k];
        }
        float dist = sqrtf(dd), ds0 = sqrtf(d0), ds1 = sqrtf(d1);
        __bf16* rb = lds_big + t * 296 + 256;
        rb[0] = (__bf16)dist;
        rb[1] = (__bf16)ds0;
        rb[2] = (__bf16)ds1;
        rb[3] = (__bf16)(p0 / (dist * ds0));
        rb[4] = (__bf16)(p1 / (dist * ds1));
        rb[5] = (__bf16)(p01 / (ds0 * ds1));
        #pragma unroll
        for (int k = 6; k < 32; ++k) rb[k] = (__bf16)0.0f;
        float* cd = s_cd + t * 6;
        cd[0] = da[0];  cd[1] = da[1];  cd[2] = da[2];
        cd[3] = cd1[0]; cd[4] = cd1[1]; cd[5] = cd1[2];
    }
    __syncthreads();

    const int mt   = w >> 1;          // edge-row tile (16 edges)
    const int nh   = w & 1;           // which half of the 128 output features
    const int arow = mt * 16 + ln;    // local edge row this lane's A-frag maps to

    // ---------------- Layer 1: [64,288] x [288,128] ----------------
    {
        v8f acc[4] = {};
        for (int kt = 0; kt < 9; ++kt) {
            v16bf a = load_a_frag(lds_big + arow * 296 + kt * 32 + g * 8);
            const __bf16* bb = We1p + ((kt * 8 + nh * 4) * 32 + lane) * 16;
            #pragma unroll
            for (int j = 0; j < 4; ++j)
                acc[j] = wmma_bf16(a, *(const v16bf*)(bb + j * 512), acc[j]);
        }
        #pragma unroll
        for (int j = 0; j < 4; ++j) {
            int   ncol = (nh * 4 + j) * 16 + ln;
            float bias = be1[ncol];
            #pragma unroll
            for (int r = 0; r < 8; ++r) {
                int mr = mt * 16 + r + 8 * g;     // D-frag: M = r + 8*g
                lds_m[mr * 136 + ncol] = (__bf16)silu_f(acc[j][r] + bias);
            }
        }
    }
    __syncthreads();

    // ---------------- Layer 2: [64,128] x [128,128], aggregate m ----------------
    {
        v8f acc[4] = {};
        for (int kt = 0; kt < 4; ++kt) {
            v16bf a = load_a_frag(lds_m + arow * 136 + kt * 32 + g * 8);
            const __bf16* bb = We2p + ((kt * 8 + nh * 4) * 32 + lane) * 16;
            #pragma unroll
            for (int j = 0; j < 4; ++j)
                acc[j] = wmma_bf16(a, *(const v16bf*)(bb + j * 512), acc[j]);
        }
        #pragma unroll
        for (int j = 0; j < 4; ++j) {
            int   ncol = (nh * 4 + j) * 16 + ln;
            float bias = be2[ncol];
            #pragma unroll
            for (int r = 0; r < 8; ++r) {
                int mr = mt * 16 + r + 8 * g;
                float sv = silu_f(acc[j][r] + bias);
                lds_big[mr * 136 + ncol] = (__bf16)sv;   // m2 (reuse e_in LDS)
                unsafeAtomicAdd(agg + (size_t)s_row[mr] * 128 + ncol, sv);
            }
        }
    }
    __syncthreads();

    // ---------------- Coord hidden: p = silu(m2 @ Wc1 + bc1) ----------------
    {
        v8f acc[4] = {};
        for (int kt = 0; kt < 4; ++kt) {
            v16bf a = load_a_frag(lds_big + arow * 136 + kt * 32 + g * 8);
            const __bf16* bb = Wc1p + ((kt * 8 + nh * 4) * 32 + lane) * 16;
            #pragma unroll
            for (int j = 0; j < 4; ++j)
                acc[j] = wmma_bf16(a, *(const v16bf*)(bb + j * 512), acc[j]);
        }
        #pragma unroll
        for (int j = 0; j < 4; ++j) {
            int   ncol = (nh * 4 + j) * 16 + ln;
            float bias = bc1[ncol];
            #pragma unroll
            for (int r = 0; r < 8; ++r) {
                int mr = mt * 16 + r + 8 * g;
                lds_m[mr * 136 + ncol] = (__bf16)silu_f(acc[j][r] + bias);
            }
        }
    }
    __syncthreads();

    // ---------------- cw = p @ Wc2 (N padded to 16), coord aggregation --------
    if (w < 4) {                       // wave-uniform: EXEC stays all-1 for WMMA
        v8f cacc = {};
        const int arowc = w * 16 + ln;
        for (int kt = 0; kt < 4; ++kt) {
            v16bf a = load_a_frag(lds_m + arowc * 136 + kt * 32 + g * 8);
            v16bf b = *(const v16bf*)(Wc2p + ((size_t)kt * 32 + lane) * 16);
            cacc = wmma_bf16(a, b, cacc);
        }
        if (ln < 2) {                  // only channels 0,1 are real
            #pragma unroll
            for (int r = 0; r < 8; ++r) {
                int   mr  = w * 16 + r + 8 * g;
                float cw  = cacc[r];
                const float* cd = s_cd + mr * 6 + ln * 3;
                int   rowi = s_row[mr];
                float* dst = aggc + (size_t)rowi * 6 + ln * 3;
                unsafeAtomicAdd(dst + 0, cd[0] * cw);
                unsafeAtomicAdd(dst + 1, cd[1] * cw);
                unsafeAtomicAdd(dst + 2, cd[2] * cw);
                if (ln == 0) unsafeAtomicAdd(cnt + rowi, 1.0f);
            }
        }
    }
}

// ---------------------------------------------------------------------------
// Node kernel: h_out = h + silu([h|agg] @ Wn1 + bn1) @ Wn2 + bn2
// 64 nodes per block, 256 threads.
// ---------------------------------------------------------------------------
__global__ __launch_bounds__(256, 1)
void egcl_node_kernel(const float* __restrict__ h,
                      const float* __restrict__ agg,
                      const float* __restrict__ bn1,
                      const float* __restrict__ bn2,
                      const __bf16* __restrict__ Wn1p,
                      const __bf16* __restrict__ Wn2p,
                      float* __restrict__ hout,
                      int Nn)
{
    __shared__ __bf16 nin[64 * 264];   // [64][256] + pad
    __shared__ __bf16 mb[64 * 136];

    const int t    = threadIdx.x;
    const int w    = t >> 5;
    const int lane = t & 31;
    const int g    = lane >> 4;
    const int ln   = lane & 15;
    const int nbase = blockIdx.x * 64;

    {   // stage n_in = [h | agg]
        int nl = t >> 2, p = t & 3;
        int node = nbase + nl;
        uint4* dst = (uint4*)(nin + nl * 264 + p * 64);
        if (node < Nn) {
            const float* srcp = (p < 2)
                ? (h   + (size_t)node * 128 + (p & 1) * 64)
                : (agg + (size_t)node * 128 + (p & 1) * 64);
            const float4* sp = (const float4*)srcp;
            #pragma unroll
            for (int i = 0; i < 8; ++i) {
                float4 x = sp[2 * i];
                float4 y = sp[2 * i + 1];
                union { __bf16 b[8]; uint4 u; } pk;
                pk.b[0] = (__bf16)x.x; pk.b[1] = (__bf16)x.y;
                pk.b[2] = (__bf16)x.z; pk.b[3] = (__bf16)x.w;
                pk.b[4] = (__bf16)y.x; pk.b[5] = (__bf16)y.y;
                pk.b[6] = (__bf16)y.z; pk.b[7] = (__bf16)y.w;
                dst[i] = pk.u;
            }
        } else {
            uint4 z = make_uint4(0u, 0u, 0u, 0u);
            #pragma unroll
            for (int i = 0; i < 8; ++i) dst[i] = z;
        }
    }
    __syncthreads();

    const int mt   = w >> 1;
    const int nh   = w & 1;
    const int arow = mt * 16 + ln;

    // ---- hidden = silu(n_in @ Wn1 + bn1) : K = 256 ----
    {
        v8f acc[4] = {};
        for (int kt = 0; kt < 8; ++kt) {
            v16bf a = load_a_frag(nin + arow * 264 + kt * 32 + g * 8);
            const __bf16* bb = Wn1p + ((kt * 8 + nh * 4) * 32 + lane) * 16;
            #pragma unroll
            for (int j = 0; j < 4; ++j)
                acc[j] = wmma_bf16(a, *(const v16bf*)(bb + j * 512), acc[j]);
        }
        #pragma unroll
        for (int j = 0; j < 4; ++j) {
            int   ncol = (nh * 4 + j) * 16 + ln;
            float bias = bn1[ncol];
            #pragma unroll
            for (int r = 0; r < 8; ++r) {
                int mr = mt * 16 + r + 8 * g;
                mb[mr * 136 + ncol] = (__bf16)silu_f(acc[j][r] + bias);
            }
        }
    }
    __syncthreads();

    // ---- h_out = h + hidden @ Wn2 + bn2 : K = 128 ----
    {
        v8f acc[4] = {};
        for (int kt = 0; kt < 4; ++kt) {
            v16bf a = load_a_frag(mb + arow * 136 + kt * 32 + g * 8);
            const __bf16* bb = Wn2p + ((kt * 8 + nh * 4) * 32 + lane) * 16;
            #pragma unroll
            for (int j = 0; j < 4; ++j)
                acc[j] = wmma_bf16(a, *(const v16bf*)(bb + j * 512), acc[j]);
        }
        #pragma unroll
        for (int j = 0; j < 4; ++j) {
            int   ncol = (nh * 4 + j) * 16 + ln;
            float bias = bn2[ncol];
            #pragma unroll
            for (int r = 0; r < 8; ++r) {
                int mr   = mt * 16 + r + 8 * g;
                int node = nbase + mr;
                if (node < Nn) {
                    size_t off = (size_t)node * 128 + ncol;
                    hout[off] = h[off] + acc[j][r] + bias;
                }
            }
        }
    }
}

// coord_out = coord + agg_c / max(cnt, 1)
__global__ void egcl_coord_kernel(const float* __restrict__ coord,
                                  const float* __restrict__ aggc,
                                  const float* __restrict__ cnt,
                                  float* __restrict__ cout, int Nn)
{
    int i = blockIdx.x * blockDim.x + threadIdx.x;
    if (i < Nn * 6) {
        int n = i / 6;
        float c = cnt[n];
        c = c < 1.0f ? 1.0f : c;
        cout[i] = coord[i] + aggc[i] / c;
    }
}

extern "C" void kernel_launch(void* const* d_in, const int* in_sizes, int n_in,
                              void* d_out, int out_size, void* d_ws, size_t ws_size,
                              hipStream_t stream) {
    const float* h     = (const float*)d_in[0];
    const int*   eidx  = (const int*)  d_in[1];
    const float* coord = (const float*)d_in[2];
    const float* We1   = (const float*)d_in[3];
    const float* be1   = (const float*)d_in[4];
    const float* We2   = (const float*)d_in[5];
    const float* be2   = (const float*)d_in[6];
    const float* Wn1   = (const float*)d_in[7];
    const float* bn1   = (const float*)d_in[8];
    const float* Wn2   = (const float*)d_in[9];
    const float* bn2   = (const float*)d_in[10];
    const float* Wc1   = (const float*)d_in[11];
    const float* bc1   = (const float*)d_in[12];
    const float* Wc2   = (const float*)d_in[13];

    const int Nn = in_sizes[0] / 128;   // 50000
    const int Ne = in_sizes[1] / 2;     // 800000

    // workspace layout
    float* agg  = (float*)d_ws;                 // [N,128]
    float* aggc = agg  + (size_t)Nn * 128;      // [N,2,3]
    float* cnt  = aggc + (size_t)Nn * 6;        // [N]
    __bf16* We1p = (__bf16*)(cnt + Nn);         // 288*128
    __bf16* We2p = We1p + 288 * 128;            // 128*128
    __bf16* Wn1p = We2p + 128 * 128;            // 256*128
    __bf16* Wn2p = Wn1p + 256 * 128;            // 128*128
    __bf16* Wc1p = Wn2p + 128 * 128;            // 128*128
    __bf16* Wc2p = Wc1p + 128 * 128;            // 128*16 (N padded)

    float* hout = (float*)d_out;                // [N,128]
    float* cout = hout + (size_t)Nn * 128;      // [N,2,3]

    // zero aggregation buffers
    long long nz = (long long)Nn * 135;
    zero_kernel<<<dim3((unsigned)((nz + 255) / 256)), dim3(256), 0, stream>>>(agg, nz);

    // pre-swizzle weights into WMMA B-fragment layout (bf16)
    pack_weight_kernel<<<dim3((288 * 128 + 255) / 256), dim3(256), 0, stream>>>(We1, We1p, 288, 262, 128, 128);
    pack_weight_kernel<<<dim3((128 * 128 + 255) / 256), dim3(256), 0, stream>>>(We2, We2p, 128, 128, 128, 128);
    pack_weight_kernel<<<dim3((256 * 128 + 255) / 256), dim3(256), 0, stream>>>(Wn1, Wn1p, 256, 256, 128, 128);
    pack_weight_kernel<<<dim3((128 * 128 + 255) / 256), dim3(256), 0, stream>>>(Wn2, Wn2p, 128, 128, 128, 128);
    pack_weight_kernel<<<dim3((128 * 128 + 255) / 256), dim3(256), 0, stream>>>(Wc1, Wc1p, 128, 128, 128, 128);
    pack_weight_kernel<<<dim3((128 * 16 + 255) / 256), dim3(256), 0, stream>>>(Wc2, Wc2p, 128, 128, 16, 2);

    // fused edge pipeline
    egcl_edge_kernel<<<dim3((Ne + 63) / 64), dim3(256), 0, stream>>>(
        h, eidx, coord, be1, be2, bc1, We1p, We2p, Wc1p, Wc2p,
        agg, aggc, cnt, Nn, Ne);

    // node MLP + residual
    egcl_node_kernel<<<dim3((Nn + 63) / 64), dim3(256), 0, stream>>>(
        h, agg, bn1, bn2, Wn1p, Wn2p, hout, Nn);

    // coord update
    egcl_coord_kernel<<<dim3((Nn * 6 + 255) / 256), dim3(256), 0, stream>>>(
        coord, aggc, cnt, cout, Nn);
}
